// Attention_29420525977701
// MI455X (gfx1250) — compile-verified, matrix operand order checked by reading, not run on hardware
//
#include <hip/hip_runtime.h>

// ---------------------------------------------------------------------------
// Problem constants: b=2, s=2048, DIM=2048, 16 heads / 4 KV heads, hd=128.
// ---------------------------------------------------------------------------
constexpr int BATCH = 2;
constexpr int SEQ   = 2048;
constexpr int DIM   = 2048;
constexpr int NH    = 16;
constexpr int NKV   = 4;
constexpr int HD    = 128;
constexpr int QKVD  = DIM + 2 * NKV * HD;   // 3072
constexpr int ROWS  = BATCH * SEQ;          // 4096

typedef __bf16 bf16_t;
typedef bf16_t v16bf __attribute__((ext_vector_type(16)));
typedef bf16_t v8bf  __attribute__((ext_vector_type(8)));
typedef float  v8f   __attribute__((ext_vector_type(8)));
typedef unsigned int u32;
typedef u32 v4u __attribute__((ext_vector_type(4)));
typedef int  v8i __attribute__((ext_vector_type(8)));
typedef int  v4i __attribute__((ext_vector_type(4)));

#if defined(__has_builtin)
#  if __has_builtin(__builtin_amdgcn_tensor_load_to_lds)
#    define USE_TDM 1
#  endif
#endif
#ifndef USE_TDM
#  define USE_TDM 0
#endif

// ---------------------------------------------------------------------------
// WMMA fragment helpers (gfx1250 wave32, 16-bit A/B lane layout):
// lane L: row/col = L&15; elems 0..7 = K koff..koff+7, elems 8..15 = K 16+koff..
// (koff = (L>=16)?8:0). A B-frag of B[k][n]=W[n][k] is a contiguous row of W.
// ---------------------------------------------------------------------------
__device__ __forceinline__ v16bf load_ab_frag(const bf16_t* __restrict__ tile,
                                              int ld, int lane) {
  const int m    = lane & 15;
  const int koff = (lane & 16) >> 1;       // 0 or 8
  const bf16_t* p = tile + (size_t)m * ld + koff;
  v8bf lo = *(const v8bf*)(p);
  v8bf hi = *(const v8bf*)(p + 16);
  v16bf f;
#pragma unroll
  for (int i = 0; i < 8; ++i) { f[i] = lo[i]; f[i + 8] = hi[i]; }
  return f;
}

__device__ __forceinline__ v8f wmma_bf16(v16bf a, v16bf b, v8f c) {
  return __builtin_amdgcn_wmma_f32_16x16x32_bf16(false, a, false, b,
                                                 (short)0, c, false, false);
}

// Same-wave LDS producer->consumer ordering (no workgroup barrier possible in
// the divergent causal loop): DS ops are in-order per wave, just drain DScnt.
__device__ __forceinline__ void wave_lds_sync() {
  __builtin_amdgcn_wave_barrier();
  asm volatile("s_wait_dscnt 0" ::: "memory");
  __builtin_amdgcn_wave_barrier();
}

#if USE_TDM
// ---------------------------------------------------------------------------
// Tensor Data Mover: 2D tile (rows x elems) of bf16, global -> LDS,
// row-major compacted in LDS. D# packing per CDNA5 ISA ch.8 (group0/1; 2D so
// groups 2/3 zero). This toolchain exposes the 6-arg builtin
// (uint32x4 g0, int32x8 g1, int32x4 g2, int32x4 g3, int32x8 extra, i32 cpol);
// the extra group is zero for a plain 2D tile. Tracked with TENSORcnt.
// ---------------------------------------------------------------------------
__device__ __forceinline__ void tdm_load_2d_bf16(const bf16_t* g, u32 lds_off,
                                                 u32 tile_w_elems, u32 tile_rows,
                                                 u32 row_stride_elems) {
  unsigned long long ga = (unsigned long long)(uintptr_t)g;
  v4u g0;
  g0[0] = 1u;                                        // count=1, user mode
  g0[1] = lds_off;                                   // lds_addr (bytes)
  g0[2] = (u32)(ga & 0xFFFFFFFFu);                   // global_addr[31:0]
  g0[3] = (u32)((ga >> 32) & 0x01FFFFFFu) | (2u << 30); // addr[56:32], type=2
  v8i g1;
  u32 tw = row_stride_elems;                         // tensor_dim0 (no OOB)
  u32 th = 1u << 20;                                 // tensor_dim1 (no OOB)
  g1[0] = 0x00010000;                                // data_size=1 (2B), mask=0
  g1[1] = (int)((tw & 0xFFFFu) << 16);               // tensor_dim0[15:0] @63:48
  g1[2] = (int)(((tw >> 16) & 0xFFFFu) | ((th & 0xFFFFu) << 16));
  g1[3] = (int)(((th >> 16) & 0xFFFFu) | ((tile_w_elems & 0xFFFFu) << 16));
  g1[4] = (int)(tile_rows & 0xFFFFu);                // tile_dim1; tile_dim2=0
  g1[5] = (int)row_stride_elems;                     // dim0_stride[31:0]
  g1[6] = 0;
  g1[7] = 0;
  v4i z4 = {0, 0, 0, 0};
  v8i z8 = {0, 0, 0, 0, 0, 0, 0, 0};
  __builtin_amdgcn_tensor_load_to_lds(g0, g1, z4, z4, z8, 0);
}
#endif

// fallback / generic cooperative tile copy: 128x32 bf16 (8 KB), 256 threads
__device__ __forceinline__ void copy_tile_128x32(bf16_t* __restrict__ dst,
                                                 const bf16_t* __restrict__ src,
                                                 int ld, int tid) {
#pragma unroll
  for (int c = tid; c < 512; c += 256) {
    const int r = c >> 2, q = c & 3;
    *(v8bf*)(dst + r * 32 + q * 8) = *(const v8bf*)(src + (size_t)r * ld + q * 8);
  }
}

// ---------------------------------------------------------------------------
// Kernel 1: f32 -> bf16 conversion
// ---------------------------------------------------------------------------
__global__ void f32_to_bf16_kernel(const float* __restrict__ in,
                                   bf16_t* __restrict__ out, int n) {
  int i = (blockIdx.x * blockDim.x + threadIdx.x) * 4;
  if (i + 4 <= n) {
    float4 v = *(const float4*)(in + i);
    out[i + 0] = (bf16_t)v.x;
    out[i + 1] = (bf16_t)v.y;
    out[i + 2] = (bf16_t)v.z;
    out[i + 3] = (bf16_t)v.w;
  } else {
    for (; i < n; ++i) out[i] = (bf16_t)in[i];
  }
}

// ---------------------------------------------------------------------------
// Kernel 2/5: GEMM  out[m][n] = sum_k A[m][k]*W[n][k]  (bf16 in, f32 acc).
// Block = 256 threads / 8 waves; block tile 128x128, K-step 32.
// LDS double-buffered 128x32 A and B tiles; TDM DMA overlapped with compute.
// Wave grid 2(M) x 4(N): each wave owns 64x32 = 4x2 WMMA tiles.
// ---------------------------------------------------------------------------
template <bool OUT_BF16>
__global__ __launch_bounds__(256)
void gemm_bf16_wmma_kernel(const bf16_t* __restrict__ A,
                           const bf16_t* __restrict__ W,
                           void* __restrict__ outp, int M, int N, int K) {
  __shared__ __align__(16) bf16_t sA[2][128 * 32];
  __shared__ __align__(16) bf16_t sB[2][128 * 32];

  const int tid  = threadIdx.x;
  const int lane = tid & 31;
  const int wave = tid >> 5;
  const int wm   = wave & 1;          // 0..1  -> 64-row half
  const int wn   = wave >> 1;         // 0..3  -> 32-col quarter
  const int row0 = blockIdx.y * 128;
  const int col0 = blockIdx.x * 128;
  const int NT   = K / 32;

  // prologue: stage tile 0 into buffer 0
#if USE_TDM
  if (wave == 0) {
    tdm_load_2d_bf16(A + (size_t)row0 * K, (u32)(uintptr_t)&sA[0][0], 32, 128, K);
    tdm_load_2d_bf16(W + (size_t)col0 * K, (u32)(uintptr_t)&sB[0][0], 32, 128, K);
  }
#else
  copy_tile_128x32(sA[0], A + (size_t)row0 * K, K, tid);
  copy_tile_128x32(sB[0], W + (size_t)col0 * K, K, tid);
#endif

  v8f acc[4][2] = {};
  for (int i = 0; i < NT; ++i) {
    __syncthreads();                       // buf[(i+1)&1] free for refill
#if USE_TDM
    if (wave == 0) {
      if (i + 1 < NT) {
        const int k1 = (i + 1) * 32;
        tdm_load_2d_bf16(A + (size_t)row0 * K + k1,
                         (u32)(uintptr_t)&sA[(i + 1) & 1][0], 32, 128, K);
        tdm_load_2d_bf16(W + (size_t)col0 * K + k1,
                         (u32)(uintptr_t)&sB[(i + 1) & 1][0], 32, 128, K);
        __builtin_amdgcn_s_wait_tensorcnt(2);   // tile i done, i+1 in flight
      } else {
        __builtin_amdgcn_s_wait_tensorcnt(0);
      }
    }
#else
    if (i + 1 < NT) {
      const int k1 = (i + 1) * 32;
      copy_tile_128x32(sA[(i + 1) & 1], A + (size_t)row0 * K + k1, K, tid);
      copy_tile_128x32(sB[(i + 1) & 1], W + (size_t)col0 * K + k1, K, tid);
    }
#endif
    __syncthreads();                       // tile i visible to all waves

    const bf16_t* tA = sA[i & 1];
    const bf16_t* tB = sB[i & 1];
    v16bf bfr[2];
#pragma unroll
    for (int nt = 0; nt < 2; ++nt)
      bfr[nt] = load_ab_frag(tB + (wn * 32 + nt * 16) * 32, 32, lane);
#pragma unroll
    for (int mt = 0; mt < 4; ++mt) {
      v16bf af = load_ab_frag(tA + (wm * 64 + mt * 16) * 32, 32, lane);
#pragma unroll
      for (int nt = 0; nt < 2; ++nt)
        acc[mt][nt] = wmma_bf16(af, bfr[nt], acc[mt][nt]);
    }
  }

  // D layout: lane holds col n = lane&15, rows m = r + 8*(lane>=16)
  const int n_l   = lane & 15;
  const int mbase = (lane & 16) >> 1;
#pragma unroll
  for (int mt = 0; mt < 4; ++mt)
#pragma unroll
    for (int nt = 0; nt < 2; ++nt)
#pragma unroll
      for (int r = 0; r < 8; ++r) {
        size_t idx = (size_t)(row0 + wm * 64 + mt * 16 + mbase + r) * N
                   + (col0 + wn * 32 + nt * 16 + n_l);
        if (OUT_BF16) ((bf16_t*)outp)[idx] = (bf16_t)acc[mt][nt][r];
        else          ((float*)outp)[idx]  = acc[mt][nt][r];
      }
}

// ---------------------------------------------------------------------------
// Kernel 3: RoPE + head split. Q is pre-scaled by 1/sqrt(hd) so the attention
// hot loop needs no per-score multiply.
//   Q -> [B][NH][SEQ][HD], K -> [B][NKV][SEQ][HD], Vt -> [B][NKV][HD][SEQ]
// ---------------------------------------------------------------------------
__global__ void rope_split_kernel(const bf16_t* __restrict__ qkv,
                                  const float* __restrict__ fc,
                                  bf16_t* __restrict__ Q,
                                  bf16_t* __restrict__ Kd,
                                  bf16_t* __restrict__ Vt) {
  const int idx = blockIdx.x * blockDim.x + threadIdx.x;
  const int PAIRS = QKVD / 2;
  if (idx >= ROWS * PAIRS) return;
  const int p     = idx % PAIRS;
  const int rowi  = idx / PAIRS;
  const int batch = rowi >> 11;
  const int sidx  = rowi & (SEQ - 1);
  const int col   = p * 2;
  const float ASCALE = 0.08838834764831845f;  // hd^-0.5 folded into Q

  const float x0 = (float)qkv[(size_t)rowi * QKVD + col];
  const float x1 = (float)qkv[(size_t)rowi * QKVD + col + 1];

  if (col < DIM) {                                  // Q (rope + scale)
    const int h = col >> 7, d = col & 127, i = d >> 1;
    const float c  = fc[(sidx * 64 + i) * 2 + 0];
    const float sn = fc[(sidx * 64 + i) * 2 + 1];
    size_t base = (((size_t)batch * NH + h) * SEQ + sidx) * HD + d;
    Q[base]     = (bf16_t)((x0 * c - x1 * sn) * ASCALE);
    Q[base + 1] = (bf16_t)((x1 * c + x0 * sn) * ASCALE);
  } else if (col < DIM + NKV * HD) {                // K (rope)
    const int cc = col - DIM;
    const int h = cc >> 7, d = cc & 127, i = d >> 1;
    const float c  = fc[(sidx * 64 + i) * 2 + 0];
    const float sn = fc[(sidx * 64 + i) * 2 + 1];
    size_t base = (((size_t)batch * NKV + h) * SEQ + sidx) * HD + d;
    Kd[base]     = (bf16_t)(x0 * c - x1 * sn);
    Kd[base + 1] = (bf16_t)(x1 * c + x0 * sn);
  } else {                                          // V (transpose, no rope)
    const int cc = col - (DIM + NKV * HD);
    const int h = cc >> 7, d = cc & 127;
    size_t base = (((size_t)batch * NKV + h) * HD + d) * SEQ + sidx;
    Vt[base]       = (bf16_t)x0;
    Vt[base + SEQ] = (bf16_t)x1;
  }
}

// ---------------------------------------------------------------------------
// Kernel 4: causal GQA flash attention; one wave per 16-row Q tile, keys in
// chunks of 32, online softmax. Masked compare code only in the 1-2 diagonal
// chunks (MASK template), unmasked steady state is pure wmma/ds/exp.
// ---------------------------------------------------------------------------
template <bool MASK>
__device__ __forceinline__ void attn_chunk(
    int kb, int q_abs, int lane, int row, int half, int mbase,
    const bf16_t* __restrict__ Kp, const bf16_t* __restrict__ Vp,
    const v16bf (&qf)[4], float* __restrict__ myS, bf16_t* __restrict__ myP,
    v8f (&o)[8], float& row_m, float& row_l) {
  // ---- S = Q K^T (two 16x16 tiles over 32 keys) -> LDS row-major
#pragma unroll
  for (int t = 0; t < 2; ++t) {
    v8f s = {};
#pragma unroll
    for (int c = 0; c < 4; ++c) {
      v16bf kf = load_ab_frag(Kp + (size_t)(kb + t * 16) * HD + c * 32, HD, lane);
      s = wmma_bf16(qf[c], kf, s);
    }
#pragma unroll
    for (int r = 0; r < 8; ++r)
      myS[(mbase + r) * 32 + t * 16 + (lane & 15)] = s[r];
  }
  wave_lds_sync();

  // ---- online softmax: lane handles (row, keys half*16 .. half*16+15)
  float vals[16];
  float mx = -3.0e38f;
#pragma unroll
  for (int j = 0; j < 16; ++j) {
    float v = myS[row * 32 + half * 16 + j];
    if (MASK) {
      const int key = kb + half * 16 + j;
      if (key > q_abs) v = -3.0e38f;
    }
    vals[j] = v;
    mx = fmaxf(mx, v);
  }
  mx = fmaxf(mx, __shfl_xor(mx, 16));
  const float m_new = fmaxf(row_m, mx);
  const float alpha = __expf(row_m - m_new);
  float lsum = 0.0f;
#pragma unroll
  for (int j = 0; j < 16; ++j) {
    const float pv = __expf(vals[j] - m_new);
    lsum += pv;
    myP[row * 32 + half * 16 + j] = (bf16_t)pv;
  }
  lsum += __shfl_xor(lsum, 16);
  row_l = row_l * alpha + lsum;
  row_m = m_new;
  wave_lds_sync();

  // ---- rescale O by per-row alpha, then O += P(16x32) * V(32x128)
  float al[8];
#pragma unroll
  for (int r = 0; r < 8; ++r) al[r] = __shfl(alpha, mbase + r);
#pragma unroll
  for (int t = 0; t < 8; ++t)
#pragma unroll
    for (int r = 0; r < 8; ++r) o[t][r] *= al[r];

  v16bf pf = load_ab_frag(myP, 32, lane);
#pragma unroll
  for (int t = 0; t < 8; ++t) {
    v16bf vf = load_ab_frag(Vp + (size_t)(t * 16) * SEQ + kb, SEQ, lane);
    o[t] = wmma_bf16(pf, vf, o[t]);
  }
}

__global__ __launch_bounds__(256)
void flash_attn_kernel(const bf16_t* __restrict__ Q,
                       const bf16_t* __restrict__ Kd,
                       const bf16_t* __restrict__ Vt,
                       bf16_t* __restrict__ Ob) {
  __shared__ __align__(16) float  sS[8][16 * 32];
  __shared__ __align__(16) bf16_t sP[8][16 * 32];

  const int lane  = threadIdx.x & 31;
  const int wave  = threadIdx.x >> 5;
  const int gwave = blockIdx.x * 8 + wave;          // 4096 total
  const int qt    = gwave & 127;
  const int head  = (gwave >> 7) & 15;
  const int batch = gwave >> 11;
  const int kvh   = head >> 2;                      // GQA
  const int q0    = qt * 16;

  const bf16_t* Qp = Q  + (((size_t)batch * NH  + head) * SEQ + q0) * HD;
  const bf16_t* Kp = Kd + ((size_t)batch * NKV + kvh) * SEQ * HD;
  const bf16_t* Vp = Vt + ((size_t)batch * NKV + kvh) * HD * SEQ;
  float*  myS = sS[wave];
  bf16_t* myP = sP[wave];

  v16bf qf[4];
#pragma unroll
  for (int c = 0; c < 4; ++c) qf[c] = load_ab_frag(Qp + c * 32, HD, lane);

  v8f o[8] = {};
  const int row   = lane & 15;
  const int half  = lane >> 4;
  const int mbase = half * 8;
  const int q_abs = q0 + row;
  float row_m = -3.0e38f, row_l = 0.0f;

  int kb = 0;
  for (; kb + 32 <= q0; kb += 32)                   // fully-unmasked chunks
    attn_chunk<false>(kb, q_abs, lane, row, half, mbase, Kp, Vp, qf,
                      myS, myP, o, row_m, row_l);
  for (; kb < q0 + 16; kb += 32)                    // diagonal (masked) chunks
    attn_chunk<true>(kb, q_abs, lane, row, half, mbase, Kp, Vp, qf,
                     myS, myP, o, row_m, row_l);

  // ---- normalize + store: out[batch*SEQ + q, head*128 + d]
  const float inv = 1.0f / row_l;
  float ir[8];
#pragma unroll
  for (int r = 0; r < 8; ++r) ir[r] = __shfl(inv, mbase + r);
  const int n_l = lane & 15;
#pragma unroll
  for (int t = 0; t < 8; ++t)
#pragma unroll
    for (int r = 0; r < 8; ++r) {
      size_t idx = (size_t)(batch * SEQ + q0 + mbase + r) * DIM
                 + head * HD + t * 16 + n_l;
      Ob[idx] = (bf16_t)(o[t][r] * ir[r]);
    }
}

// ---------------------------------------------------------------------------
// Workspace layout (~100 MiB); every region fully written before read.
// ---------------------------------------------------------------------------
constexpr size_t OFF_XB   = 0;
constexpr size_t SZ_XB    = (size_t)ROWS * DIM * 2;
constexpr size_t OFF_WQKV = OFF_XB + SZ_XB;
constexpr size_t SZ_WQKV  = (size_t)QKVD * DIM * 2;
constexpr size_t OFF_WO   = OFF_WQKV + SZ_WQKV;
constexpr size_t SZ_WO    = (size_t)DIM * DIM * 2;
constexpr size_t OFF_QKV  = OFF_WO + SZ_WO;
constexpr size_t SZ_QKV   = (size_t)ROWS * QKVD * 2;
constexpr size_t OFF_Q    = OFF_QKV + SZ_QKV;
constexpr size_t SZ_Q     = (size_t)BATCH * NH * SEQ * HD * 2;
constexpr size_t OFF_K    = OFF_Q + SZ_Q;
constexpr size_t SZ_K     = (size_t)BATCH * NKV * SEQ * HD * 2;
constexpr size_t OFF_VT   = OFF_K + SZ_K;
constexpr size_t SZ_VT    = SZ_K;
constexpr size_t OFF_AO   = OFF_VT + SZ_VT;

extern "C" void kernel_launch(void* const* d_in, const int* in_sizes, int n_in,
                              void* d_out, int out_size, void* d_ws, size_t ws_size,
                              hipStream_t stream) {
  const float* x     = (const float*)d_in[0];   // [2,2048,2048]
  const float* freqs = (const float*)d_in[1];   // [2048,64,2]
  const float* wqkv  = (const float*)d_in[2];   // [3072,2048]
  const float* wo    = (const float*)d_in[3];   // [2048,2048]
  float* out = (float*)d_out;                   // [2,2048,2048]

  char* ws = (char*)d_ws;
  bf16_t* xb    = (bf16_t*)(ws + OFF_XB);
  bf16_t* wqkvb = (bf16_t*)(ws + OFF_WQKV);
  bf16_t* wob   = (bf16_t*)(ws + OFF_WO);
  bf16_t* qkvb  = (bf16_t*)(ws + OFF_QKV);
  bf16_t* qb    = (bf16_t*)(ws + OFF_Q);
  bf16_t* kbuf  = (bf16_t*)(ws + OFF_K);
  bf16_t* vtb   = (bf16_t*)(ws + OFF_VT);
  bf16_t* aob   = (bf16_t*)(ws + OFF_AO);

  // 1) one-time bf16 conversion
  {
    int n = ROWS * DIM;
    f32_to_bf16_kernel<<<(n / 4 + 255) / 256, 256, 0, stream>>>(x, xb, n);
    n = QKVD * DIM;
    f32_to_bf16_kernel<<<(n / 4 + 255) / 256, 256, 0, stream>>>(wqkv, wqkvb, n);
    n = DIM * DIM;
    f32_to_bf16_kernel<<<(n / 4 + 255) / 256, 256, 0, stream>>>(wo, wob, n);
  }

  // 2) QKV projection -> [4096,3072] bf16
  gemm_bf16_wmma_kernel<true>
      <<<dim3(QKVD / 128, ROWS / 128), 256, 0, stream>>>(
          xb, wqkvb, qkvb, ROWS, QKVD, DIM);

  // 3) RoPE + split (Q pre-scaled by hd^-0.5)
  {
    int n = ROWS * (QKVD / 2);
    rope_split_kernel<<<(n + 255) / 256, 256, 0, stream>>>(
        qkvb, freqs, qb, kbuf, vtb);
  }

  // 4) causal GQA flash attention
  flash_attn_kernel<<<BATCH * NH * (SEQ / 16) / 8, 256, 0, stream>>>(
      qb, kbuf, vtb, aob);

  // 5) output projection -> d_out f32
  gemm_bf16_wmma_kernel<false>
      <<<dim3(DIM / 128, ROWS / 128), 256, 0, stream>>>(
          aob, wob, out, ROWS, DIM, DIM);
}
